// Attn_30709016166860
// MI455X (gfx1250) — matrix operand order repaired and checked by reading, not compiled, not run on hardware
//
#include <hip/hip_runtime.h>
#include <hip/hip_bf16.h>

typedef __attribute__((ext_vector_type(16))) __bf16 v16bf;
typedef __attribute__((ext_vector_type(2)))  __bf16 v2bf;
typedef __attribute__((ext_vector_type(8)))  float  v8f;
typedef __attribute__((ext_vector_type(4)))  float  f4v;
typedef __attribute__((ext_vector_type(4)))  unsigned u32x4;
typedef __attribute__((ext_vector_type(8)))  int      i32x8;
typedef __attribute__((ext_vector_type(4)))  int      i32x4;

#if __has_builtin(__builtin_amdgcn_tanhf)
#define TANHF(x) __builtin_amdgcn_tanhf(x)
#elif __has_builtin(__builtin_amdgcn_tanh_f32)
#define TANHF(x) __builtin_amdgcn_tanh_f32(x)
#else
#define TANHF(x) tanhf(x)
#endif

#if __has_builtin(__builtin_amdgcn_tensor_load_to_lds)
#define HAVE_TDM 1
#endif

#if __has_builtin(__builtin_amdgcn_s_wait_tensorcnt)
#define WAIT_TENSOR() __builtin_amdgcn_s_wait_tensorcnt(0)
#else
#define WAIT_TENSOR() asm volatile("s_wait_tensorcnt 0x0" ::: "memory")
#endif

#define Hdim   1024
#define BATCH  32
#define SEQ    2048
#define MROWS  65536          // SEQ * BATCH
#define MT     128            // rows per block
#define NT     128            // cols per block N-panel (8 wmma subtiles / wave)
#define KC     32             // K staged per iteration (one bf16 WMMA K-step)
#define LSTR   20             // LDS row stride in dwords (16 data + 4 pad, 16B aligned)
#define LBUF   (128 * LSTR)   // one LDS buffer in dwords (10240 B)
#define NPANEL (Hdim / NT)    // 8
#define NKS    (Hdim / KC)    // 32
#define NITER  (NPANEL * NKS) // 256

// f32 pair -> packed bf16 dword (RNE). Only used in the one-pass convert kernels.
__device__ __forceinline__ unsigned pk_bf16(float a, float b) {
#if __has_builtin(__builtin_amdgcn_cvt_pk_bf16_f32)
    union { v2bf v; unsigned u; } cv;
    cv.v = __builtin_amdgcn_cvt_pk_bf16_f32(a, b);
    return cv.u;
#else
    unsigned ua = __float_as_uint(a), ub = __float_as_uint(b);
    unsigned ra = (ua + 0x7FFFu + ((ua >> 16) & 1u)) >> 16;
    unsigned rb = (ub + 0x7FFFu + ((ub >> 16) & 1u)) >> 16;
    return (ra & 0xFFFFu) | (rb << 16);
#endif
}

#ifdef HAVE_TDM
// Issue a TDM 2-D tile load: tile = 32 bf16 (64 B) x 128 rows, row stride
// `stride_elems` bf16, into LDS at lds_byte_addr with 16 B of padding inserted
// after every 64 B (pad_interval=3 -> 8 qwords, pad_amount=3 -> 4 dwords),
// matching the LSTR=20-dword LDS row pitch the WMMA fragment readers use.
__device__ __forceinline__ void tdm_load_2d(const void* gaddr,
                                            unsigned lds_byte_addr,
                                            unsigned tensor_d1,
                                            unsigned stride_elems) {
    u32x4 g0;
    size_t ga = (size_t)gaddr;
    g0[0] = 1u;                                    // count=1 (valid descriptor)
    g0[1] = lds_byte_addr;                         // lds_addr
    g0[2] = (unsigned)ga;                          // global_addr[31:0]
    g0[3] = (unsigned)(ga >> 32) | 0x80000000u;    // global_addr[56:32] | type=2
    i32x8 g1;
    g1[0] = (int)((1u << 16)      // data_size = 1 -> 2 bytes (bf16)
                | (1u << 20)      // pad_enable
                | (3u << 22)      // pad_interval: every 16 dwords
                | (3u << 25));    // pad_amount: 4 dwords
    g1[1] = (int)((unsigned)Hdim << 16);                       // tensor_dim0 lo16
    g1[2] = (int)(((unsigned)Hdim >> 16) | ((tensor_d1 & 0xFFFFu) << 16));
    g1[3] = (int)((tensor_d1 >> 16) | (32u << 16));            // tile_dim0 = 32
    g1[4] = 128;                                               // tile_dim1 = 128
    g1[5] = (int)stride_elems;                                 // dim0 stride lo32
    g1[6] = 0;
    g1[7] = 0;
    i32x4 z4 = {0, 0, 0, 0};
    i32x8 z8 = {0, 0, 0, 0, 0, 0, 0, 0};
    __builtin_amdgcn_tensor_load_to_lds(g0, g1, z4, z4, z8, 0);
}
#endif

// ---------------------------------------------------------------------------
// Kernel 0a: one-pass convert enc (f32 [M,H]) -> encb (bf16 [M,H])
// ---------------------------------------------------------------------------
__global__ void cvt_enc_kernel(const float* __restrict__ enc,
                               unsigned* __restrict__ encb_dw) {
    const size_t gid = (size_t)blockIdx.x * 256 + threadIdx.x;   // 16 floats each
    const f4v* src = (const f4v*)(enc + gid * 16);
    f4v f0 = __builtin_nontemporal_load(src + 0);
    f4v f1 = __builtin_nontemporal_load(src + 1);
    f4v f2 = __builtin_nontemporal_load(src + 2);
    f4v f3 = __builtin_nontemporal_load(src + 3);
    uint4* dst = (uint4*)(encb_dw + gid * 8);
    dst[0] = uint4{pk_bf16(f0.x, f0.y), pk_bf16(f0.z, f0.w),
                   pk_bf16(f1.x, f1.y), pk_bf16(f1.z, f1.w)};
    dst[1] = uint4{pk_bf16(f2.x, f2.y), pk_bf16(f2.z, f2.w),
                   pk_bf16(f3.x, f3.y), pk_bf16(f3.z, f3.w)};
}

// ---------------------------------------------------------------------------
// Kernel 0b: extract We half of W -> Wb (bf16 [N=1024][K=1024], K-major)
// ---------------------------------------------------------------------------
__global__ void cvt_w_kernel(const float* __restrict__ W,
                             unsigned* __restrict__ wb_dw) {
    const size_t gid = (size_t)blockIdx.x * 256 + threadIdx.x;   // 16 floats each
    const int row = (int)(gid >> 6);
    const int col = ((int)gid & 63) * 16;
    const float4* src = (const float4*)(W + (size_t)row * (2 * Hdim) + Hdim + col);
    float4 f0 = src[0], f1 = src[1], f2 = src[2], f3 = src[3];
    uint4* dst = (uint4*)(wb_dw + gid * 8);
    dst[0] = uint4{pk_bf16(f0.x, f0.y), pk_bf16(f0.z, f0.w),
                   pk_bf16(f1.x, f1.y), pk_bf16(f1.z, f1.w)};
    dst[1] = uint4{pk_bf16(f2.x, f2.y), pk_bf16(f2.z, f2.w),
                   pk_bf16(f3.x, f3.y), pk_bf16(f3.z, f3.w)};
}

// ---------------------------------------------------------------------------
// Kernel 1: hwb[b][h] = hidden[b,:] . W[h, 0:H] + bias[h]   (32x1024, tiny)
// ---------------------------------------------------------------------------
__global__ void hw_kernel(const float* __restrict__ hidden,
                          const float* __restrict__ W,
                          const float* __restrict__ bias,
                          float* __restrict__ hwb) {
    int idx = blockIdx.x * 256 + threadIdx.x;   // 0..32767
    int b = idx >> 10;
    int h = idx & (Hdim - 1);
    const float* hr = hidden + (size_t)b * Hdim;
    const float* wr = W + (size_t)h * (2 * Hdim);
    float s = 0.f;
    for (int k = 0; k < Hdim; k += 4) {
        float4 a = *(const float4*)(hr + k);
        float4 w = *(const float4*)(wr + k);
        s += a.x * w.x + a.y * w.y + a.z * w.z + a.w * w.w;
    }
    hwb[idx] = s + bias[h];
}

// ---------------------------------------------------------------------------
// Kernel 2: fused scores. TDM (or manual) staged, double-buffered bf16 WMMA.
// ---------------------------------------------------------------------------
__global__ void __launch_bounds__(256)
attn_energy_kernel(const unsigned short* __restrict__ encb,
                   const unsigned short* __restrict__ Wb,
                   const float* __restrict__ hwb,
                   const float* __restrict__ v,
                   float* __restrict__ scores_t) {
    __shared__ __align__(16) unsigned Als[2 * LBUF];   // A: 128 rows x 32 K bf16
    __shared__ __align__(16) unsigned Bsm[2 * LBUF];   // B: 128 cols x 32 K bf16

    const int t    = threadIdx.x;
    const int wid  = t >> 5;
    const int lane = t & 31;
    const int lh   = lane >> 4;          // K-group select within fragment
    const int lr   = lane & 15;          // row/col within fragment
    const int m0   = blockIdx.x * MT;

    const int srow  = t >> 1;            // staging row (manual path)
    const int shalf = t & 1;

    union FB { unsigned u[8]; uint4 q[2]; v16bf v; };

    float part[8];
#pragma unroll
    for (int k = 0; k < 8; ++k) part[k] = 0.f;

    v8f acc[8];
#pragma unroll
    for (int tt = 0; tt < 8; ++tt)
#pragma unroll
        for (int k = 0; k < 8; ++k) acc[tt][k] = 0.f;

#ifdef HAVE_TDM
    const unsigned alsBase = (unsigned)(size_t)&Als[0];
    const unsigned bsmBase = (unsigned)(size_t)&Bsm[0];
    // ---- prologue: DMA chunk 0 into buffer 0 (wave 0 only) ----
    if (wid == 0) {
        tdm_load_2d(encb + (size_t)m0 * Hdim, alsBase, MROWS, Hdim);
        tdm_load_2d(Wb, bsmBase, Hdim, Hdim);
    }
#else
    // ---- prologue: stage chunk 0 manually into buffer 0 ----
    {
        const uint4* asrc = (const uint4*)(encb + (size_t)(m0 + srow) * Hdim + shalf * 16);
        const uint4* bsrc = (const uint4*)(Wb + (size_t)srow * Hdim + shalf * 16);
        uint4 a0 = asrc[0], a1 = asrc[1];
        uint4 b0 = bsrc[0], b1 = bsrc[1];
        uint4* ad = (uint4*)&Als[srow * LSTR + shalf * 8];
        uint4* bd = (uint4*)&Bsm[srow * LSTR + shalf * 8];
        ad[0] = a0; ad[1] = a1;
        bd[0] = b0; bd[1] = b1;
    }
#endif

    int p = 0;
#ifndef HAVE_TDM
    uint4 pa0, pa1, pb0, pb1;
#endif

    for (int it = 0; it < NITER; ++it) {
        const int j = it >> 5;           // N-panel index
        const bool more = (it + 1) < NITER;
        const int jn  = (it + 1) >> 5;
        const int kcn = ((it + 1) & 31) * KC;

#ifdef HAVE_TDM
        if (wid == 0) WAIT_TENSOR();     // chunk `it` fully in LDS (issuing wave)
        __syncthreads();                 // visible to all waves; p^1 free to fill
        if (more && wid == 0) {
            const unsigned boff = (unsigned)((p ^ 1) * LBUF * 4);
            tdm_load_2d(encb + (size_t)m0 * Hdim + kcn, alsBase + boff, MROWS, Hdim);
            tdm_load_2d(Wb + (size_t)(jn * NT) * Hdim + kcn, bsmBase + boff, Hdim, Hdim);
        }
#else
        __syncthreads();
        if (more) {
            const uint4* asrc = (const uint4*)(encb + (size_t)(m0 + srow) * Hdim + kcn + shalf * 16);
            const uint4* bsrc = (const uint4*)(Wb + (size_t)(jn * NT + srow) * Hdim + kcn + shalf * 16);
            pa0 = asrc[0]; pa1 = asrc[1];
            pb0 = bsrc[0]; pb1 = bsrc[1];
        }
#endif

        // ---- compute: 1 A-fragment reused across 8 B-fragments / WMMAs ----
        {
            FB af;
            const unsigned* ar = &Als[p * LBUF + (wid * 16 + lr) * LSTR];
            af.q[0] = *(const uint4*)&ar[4 * lh];        // K = 8lh .. 8lh+7
            af.q[1] = *(const uint4*)&ar[8 + 4 * lh];    // K = 16+8lh .. 16+8lh+7
#pragma unroll
            for (int tt = 0; tt < 8; ++tt) {
                FB bf;
                const unsigned* br = &Bsm[p * LBUF + (tt * 16 + lr) * LSTR];
                bf.q[0] = *(const uint4*)&br[4 * lh];
                bf.q[1] = *(const uint4*)&br[8 + 4 * lh];
                acc[tt] = __builtin_amdgcn_wmma_f32_16x16x32_bf16(
                    false, af.v, false, bf.v, (short)0, acc[tt], false, false);
            }
        }

#ifndef HAVE_TDM
        if (more) {
            uint4* ad = (uint4*)&Als[(p ^ 1) * LBUF + srow * LSTR + shalf * 8];
            uint4* bd = (uint4*)&Bsm[(p ^ 1) * LBUF + srow * LSTR + shalf * 8];
            ad[0] = pa0; ad[1] = pa1;
            bd[0] = pb0; bd[1] = pb1;
        }
#endif
        p ^= 1;

        // ---- end of a K sweep: fused epilogue for panel j ----
        if ((it & 31) == 31) {
#pragma unroll
            for (int tt = 0; tt < 8; ++tt) {
                const int col = j * NT + tt * 16 + lr;
                const float vc = v[col];
#pragma unroll
                for (int k = 0; k < 8; ++k) {
                    const int rl   = k + 8 * lh;
                    const int bidx = (wid * 16 + rl) & (BATCH - 1);
                    float e = TANHF(acc[tt][k] + hwb[bidx * Hdim + col]);
                    part[k] += e * vc;
                }
#pragma unroll
                for (int k = 0; k < 8; ++k) acc[tt][k] = 0.f;
            }
        }
    }

    // butterfly-reduce the 16-wide row dot products within each half-wave
#pragma unroll
    for (int off = 1; off < 16; off <<= 1)
#pragma unroll
        for (int k = 0; k < 8; ++k)
            part[k] += __shfl_xor(part[k], off, 32);

    if (lr == 0) {
#pragma unroll
        for (int k = 0; k < 8; ++k) {
            int m = m0 + wid * 16 + 8 * lh + k;   // m = s*BATCH + b
            int s = m >> 5, b = m & (BATCH - 1);
            scores_t[(size_t)b * SEQ + s] = part[k];
        }
    }
}

// ---------------------------------------------------------------------------
// Kernel 3: softmax over S per batch row, in-place on d_out (block-local row)
// ---------------------------------------------------------------------------
__global__ void softmax_kernel(float* __restrict__ data) {
    __shared__ float red[256];
    const int b = blockIdx.x, t = threadIdx.x;
    float* row = data + (size_t)b * SEQ;

    float m = -3.4e38f;
    for (int s = t; s < SEQ; s += 256) m = fmaxf(m, row[s]);
    red[t] = m; __syncthreads();
    for (int o = 128; o > 0; o >>= 1) { if (t < o) red[t] = fmaxf(red[t], red[t + o]); __syncthreads(); }
    const float mx = red[0]; __syncthreads();

    float sum = 0.f;
    for (int s = t; s < SEQ; s += 256) sum += __expf(row[s] - mx);
    red[t] = sum; __syncthreads();
    for (int o = 128; o > 0; o >>= 1) { if (t < o) red[t] += red[t + o]; __syncthreads(); }
    const float inv = 1.0f / red[0];

    for (int s = t; s < SEQ; s += 256) row[s] = __expf(row[s] - mx) * inv;
}

// ---------------------------------------------------------------------------
extern "C" void kernel_launch(void* const* d_in, const int* in_sizes, int n_in,
                              void* d_out, int out_size, void* d_ws, size_t ws_size,
                              hipStream_t stream) {
    const float* hidden = (const float*)d_in[0];   // [32,1024]
    const float* enc    = (const float*)d_in[1];   // [2048,32,1024]
    const float* W      = (const float*)d_in[2];   // [1024,2048]
    const float* bias   = (const float*)d_in[3];   // [1024]
    const float* v      = (const float*)d_in[4];   // [1,1024]
    float* out = (float*)d_out;                    // [32,1,2048]

    // workspace layout (all 16B aligned):
    //   [0, 128KB)            hwb   : f32 [32][1024]
    //   [128KB, 128KB+2MB)    Wb    : bf16 [1024][1024]  (We^T, K-major)
    //   [+, +128MB)           encb  : bf16 [65536][1024]
    char* ws = (char*)d_ws;
    float*          hwb  = (float*)ws;
    unsigned*       wbdw = (unsigned*)(ws + 131072);
    unsigned*       ebdw = (unsigned*)(ws + 131072 + 2097152);
    const unsigned short* Wb   = (const unsigned short*)wbdw;
    const unsigned short* encb = (const unsigned short*)ebdw;

    cvt_enc_kernel<<<(MROWS * Hdim) / (256 * 16), 256, 0, stream>>>(enc, ebdw);
    cvt_w_kernel<<<(Hdim * Hdim) / (256 * 16), 256, 0, stream>>>(W, wbdw);
    hw_kernel<<<(BATCH * Hdim) / 256, 256, 0, stream>>>(hidden, W, bias, hwb);
    attn_energy_kernel<<<MROWS / MT, 256, 0, stream>>>(encb, Wb, hwb, v, out);
    softmax_kernel<<<BATCH, 256, 0, stream>>>(out);
}
